// ScaledDotProductAttention_70153995813465
// MI455X (gfx1250) — matrix operand order repaired and checked by reading, not compiled, not run on hardware
//
#include <hip/hip_runtime.h>

typedef _Float16 v16h __attribute__((ext_vector_type(16)));
typedef _Float16 v8h  __attribute__((ext_vector_type(8)));
typedef _Float16 v4h  __attribute__((ext_vector_type(4)));
typedef float    v8f  __attribute__((ext_vector_type(8)));
typedef float    v4f  __attribute__((ext_vector_type(4)));

#define S_LEN 2048
#define DHEAD 64
#define NEG_BIG (-3.0e38f)
// scale 1/sqrt(64) folded with log2(e): softmax computed in log2 domain (exp2 = native v_exp_f32)
#define QSCALE (0.125f * 1.44269504088896340736f)

// ============================ preprocessing ============================

// Kh[i] = (f16) K[i]   (flat, vectorized 4 floats -> 4 halfs per thread)
__global__ __launch_bounds__(256)
void prep_k_kernel(const float* __restrict__ K, _Float16* __restrict__ Kh) {
    const size_t i = ((size_t)blockIdx.x * 256 + threadIdx.x) * 4;
    v4f f = *(const v4f*)(K + i);
    v4h h;
#pragma unroll
    for (int e = 0; e < 4; ++e) h[e] = (_Float16)f[e];
    *(v4h*)(Kh + i) = h;
}

// Vt[b][d][s] = (f16) V[b][s][d]  -- LDS-tiled 64x64 transpose, coalesced both sides
__global__ __launch_bounds__(256)
void prep_vt_kernel(const float* __restrict__ V, _Float16* __restrict__ Vt) {
    __shared__ __align__(16) _Float16 T[64 * 72];   // padded stride: conflict-free
    const int b  = blockIdx.y;
    const int s0 = blockIdx.x * 64;
    const int t  = threadIdx.x;
    {
        const int sl = t >> 2, dseg = (t & 3) * 16;
        const float* vrow = V + ((size_t)b * S_LEN + s0 + sl) * DHEAD + dseg;
#pragma unroll
        for (int q = 0; q < 4; ++q) {
            v4f f = *(const v4f*)(vrow + q * 4);
#pragma unroll
            for (int e = 0; e < 4; ++e)
                T[sl * 72 + dseg + q * 4 + e] = (_Float16)f[e];
        }
    }
    __syncthreads();
    {
        const int dl = t >> 2, sseg = (t & 3) * 16;
        _Float16* orow = Vt + ((size_t)b * DHEAD + dl) * S_LEN + s0 + sseg;
        v8h a, c;
#pragma unroll
        for (int e = 0; e < 8; ++e) a[e] = T[(sseg + e) * 72 + dl];
#pragma unroll
        for (int e = 0; e < 8; ++e) c[e] = T[(sseg + 8 + e) * 72 + dl];
        *(v8h*)orow       = a;
        *(v8h*)(orow + 8) = c;
    }
}

// ============================ fast attention ============================
// 8 waves/block, 16 query rows per wave, 64-key tiles, K/V pre-converted to f16.
// No barriers: waves are fully independent (per-wave causal tile count).
// Row-sum kept as per-lane partials (lane-uniform alpha) -> reduced once at epilogue.

#define PS 72   // P LDS row stride (halfwords): 8B-aligned runs, conflict-free

template<bool NEED_MASK>
__device__ __forceinline__
void attn_tile(const _Float16* __restrict__ Kh, const _Float16* __restrict__ Vt,
               _Float16* __restrict__ PT, const v16h aq[2],
               int j, int q0, int half, int l16,
               size_t kbase, size_t vtbase,
               v8f o[4], float mr[8], float lr[8])
{
    const int kbB = half ? 8 : 0;    // B-frag k base per half-wave
    const int kbA = half ? 8 : 0;    // A-frag k base per half-wave

    // ---- S = Q @ K_j^T : 16 x 64 (4 C-frags), contraction 64 = 2 x 32 ----
    v8f s[4];
#pragma unroll
    for (int nt = 0; nt < 4; ++nt)
#pragma unroll
        for (int r = 0; r < 8; ++r) s[nt][r] = 0.0f;

#pragma unroll
    for (int nt = 0; nt < 4; ++nt) {
        const int key = j * 64 + nt * 16 + l16;
        const _Float16* krow = Kh + kbase + (size_t)key * DHEAD + kbB;
#pragma unroll
        for (int dc = 0; dc < 2; ++dc) {
            v8h h0 = *(const v8h*)(krow + dc * 32);
            v8h h1 = *(const v8h*)(krow + dc * 32 + 16);
            v16h bk;
#pragma unroll
            for (int e = 0; e < 8; ++e) { bk[e] = h0[e]; bk[8 + e] = h1[e]; }
            s[nt] = __builtin_amdgcn_wmma_f32_16x16x32_f16(
                false, aq[dc], false, bk, (short)0, s[nt], false, false);
        }
    }

    // ---- causal mask (diagonal tile only; Q pre-scaled so no mul here) ----
    if (NEED_MASK) {
#pragma unroll
        for (int nt = 0; nt < 4; ++nt) {
            const int key = j * 64 + nt * 16 + l16;
#pragma unroll
            for (int r = 0; r < 8; ++r) {
                const int row = q0 + r + half * 8;
                if (key > row) s[nt][r] = NEG_BIG;
            }
        }
    }

    // ---- online softmax (log2 domain): row max across 16 column-lanes ----
    float mnew[8], alpha[8];
#pragma unroll
    for (int r = 0; r < 8; ++r) {
        float rm = fmaxf(fmaxf(s[0][r], s[1][r]), fmaxf(s[2][r], s[3][r]));
#pragma unroll
        for (int off = 1; off < 16; off <<= 1)
            rm = fmaxf(rm, __shfl_xor(rm, off, 32));
        mnew[r]  = fmaxf(mr[r], rm);
        alpha[r] = exp2f(mr[r] - mnew[r]);
        mr[r]    = mnew[r];
    }

    // ---- P = exp2(S - m) -> per-wave LDS tile; per-lane partial row sums ----
#pragma unroll
    for (int r = 0; r < 8; ++r) {
        float rs = 0.0f;
        _Float16* prow = PT + (r + half * 8) * PS + l16;
#pragma unroll
        for (int nt = 0; nt < 4; ++nt) {
            float p = exp2f(s[nt][r] - mnew[r]);
            prow[nt * 16] = (_Float16)p;
            rs += p;
        }
        lr[r] = alpha[r] * lr[r] + rs;   // no shuffle: reduced at epilogue
    }

    // ---- rescale O ----
#pragma unroll
    for (int t = 0; t < 4; ++t)
#pragma unroll
        for (int r = 0; r < 8; ++r) o[t][r] *= alpha[r];

    // same-wave DS store->load ordering + compiler fence
    asm volatile("s_wait_dscnt 0" ::: "memory");

    // ---- reload P as two A-fragments (16x32 each) ----
    v16h ap[2];
#pragma unroll
    for (int dc = 0; dc < 2; ++dc) {
#pragma unroll
        for (int g = 0; g < 2; ++g) {
            v4h h0 = *(const v4h*)(PT + l16 * PS + dc * 32 + kbA + g * 16);
            v4h h1 = *(const v4h*)(PT + l16 * PS + dc * 32 + kbA + g * 16 + 4);
#pragma unroll
            for (int e = 0; e < 4; ++e) {
                ap[dc][g * 8 + e]     = h0[e];
                ap[dc][g * 8 + 4 + e] = h1[e];
            }
        }
    }

    // ---- O += P @ V_j : B-frags contiguous from pre-transposed f16 V ----
#pragma unroll
    for (int t = 0; t < 4; ++t) {
        const _Float16* vrow = Vt + vtbase + (size_t)(t * 16 + l16) * S_LEN + j * 64 + kbB;
#pragma unroll
        for (int dc = 0; dc < 2; ++dc) {
            v8h h0 = *(const v8h*)(vrow + dc * 32);
            v8h h1 = *(const v8h*)(vrow + dc * 32 + 16);
            v16h bv;
#pragma unroll
            for (int e = 0; e < 8; ++e) { bv[e] = h0[e]; bv[8 + e] = h1[e]; }
            o[t] = __builtin_amdgcn_wmma_f32_16x16x32_f16(
                false, ap[dc], false, bv, (short)0, o[t], false, false);
        }
    }
}

__global__ __launch_bounds__(256)
void fa_fast_kernel(const float* __restrict__ Q, const _Float16* __restrict__ Kh,
                    const _Float16* __restrict__ Vt, float* __restrict__ Out)
{
    __shared__ __align__(16) _Float16 lds[8 * 16 * PS];   // per-wave P tiles

    const int tid  = threadIdx.x;
    const int wave = tid >> 5;
    const int lane = tid & 31;
    const int half = lane >> 4;
    const int l16  = lane & 15;

    _Float16* PT = lds + wave * 16 * PS;

    const int batch = blockIdx.y;
    const int q0    = blockIdx.x * 128 + wave * 16;

    const size_t fbase  = (size_t)batch * S_LEN * DHEAD;   // Q/Out/Kh base
    const size_t vtbase = (size_t)batch * DHEAD * S_LEN;   // Vt base

    // ---- Q -> two A-fragments, scale*log2e folded in ----
    v16h aq[2];
    {
        const int kb = half ? 8 : 0;
        const float* qrow = Q + fbase + (size_t)(q0 + l16) * DHEAD;
#pragma unroll
        for (int dc = 0; dc < 2; ++dc)
#pragma unroll
            for (int g = 0; g < 2; ++g) {
                v4f f0 = *(const v4f*)(qrow + dc * 32 + kb + g * 16);
                v4f f1 = *(const v4f*)(qrow + dc * 32 + kb + g * 16 + 4);
#pragma unroll
                for (int e = 0; e < 4; ++e) {
                    aq[dc][g * 8 + e]     = (_Float16)(f0[e] * QSCALE);
                    aq[dc][g * 8 + 4 + e] = (_Float16)(f1[e] * QSCALE);
                }
            }
    }

    v8f o[4];
#pragma unroll
    for (int t = 0; t < 4; ++t)
#pragma unroll
        for (int r = 0; r < 8; ++r) o[t][r] = 0.0f;
    float mr[8], lr[8];
#pragma unroll
    for (int r = 0; r < 8; ++r) { mr[r] = NEG_BIG; lr[r] = 0.0f; }

    // per-wave causal tiles; exactly the last one straddles the diagonal
    const int ntiles = (q0 + 15) / 64 + 1;
    int j = 0;
    for (; j < ntiles - 1; ++j)
        attn_tile<false>(Kh, Vt, PT, aq, j, q0, half, l16, fbase, vtbase, o, mr, lr);
    attn_tile<true>(Kh, Vt, PT, aq, j, q0, half, l16, fbase, vtbase, o, mr, lr);

    // ---- epilogue: one-time denominator reduction, normalize, store ----
#pragma unroll
    for (int r = 0; r < 8; ++r) {
        float ls = lr[r];
#pragma unroll
        for (int off = 1; off < 16; off <<= 1)
            ls += __shfl_xor(ls, off, 32);
        const float inv = 1.0f / ls;
        float* orow = Out + fbase + (size_t)(q0 + r + half * 8) * DHEAD + l16;
#pragma unroll
        for (int t = 0; t < 4; ++t)
            orow[t * 16] = o[t][r] * inv;
    }
}

// ============================ fallback (round-1 kernel, no workspace) ============================

#define BMF 64
#define BNF 32
#define VTSF 36
#define PSF  36

__global__ __launch_bounds__(128)
void fa_causal_fallback(const float* __restrict__ Q, const float* __restrict__ K,
                        const float* __restrict__ V, float* __restrict__ Out)
{
    __shared__ __align__(16) _Float16 lds[DHEAD * VTSF + 4 * 16 * PSF];
    _Float16* VT = lds;
    const int tid  = threadIdx.x;
    const int wave = tid >> 5;
    const int lane = tid & 31;
    const int half = lane >> 4;
    const int l16  = lane & 15;
    _Float16* PT = lds + DHEAD * VTSF + wave * 16 * PSF;

    const int batch = blockIdx.y;
    const int qtile = blockIdx.x;
    const int q0    = qtile * BMF + wave * 16;
    const size_t bbase = (size_t)batch * S_LEN * DHEAD;

    v16h aq[2];
    {
        const int kb = half ? 8 : 0;
        const float* qrow = Q + bbase + (size_t)(q0 + l16) * DHEAD;
#pragma unroll
        for (int dc = 0; dc < 2; ++dc)
#pragma unroll
            for (int g = 0; g < 2; ++g) {
                v4f f0 = *(const v4f*)(qrow + dc * 32 + kb + g * 16);
                v4f f1 = *(const v4f*)(qrow + dc * 32 + kb + g * 16 + 4);
#pragma unroll
                for (int e = 0; e < 4; ++e) {
                    aq[dc][g * 8 + e]     = (_Float16)f0[e];
                    aq[dc][g * 8 + 4 + e] = (_Float16)f1[e];
                }
            }
    }

    v8f o[4];
#pragma unroll
    for (int t = 0; t < 4; ++t)
#pragma unroll
        for (int r = 0; r < 8; ++r) o[t][r] = 0.0f;
    float mr[8], lr[8];
#pragma unroll
    for (int r = 0; r < 8; ++r) { mr[r] = NEG_BIG; lr[r] = 0.0f; }

    const int ntiles = (qtile * BMF + BMF - 1) / BNF + 1;
    for (int j = 0; j < ntiles; ++j) {
        __syncthreads();
        {
            const int kk  = tid >> 2;
            const int seg = (tid & 3) * 16;
            const float* vrow = V + bbase + (size_t)(j * BNF + kk) * DHEAD + seg;
#pragma unroll
            for (int q = 0; q < 4; ++q) {
                v4f f = *(const v4f*)(vrow + q * 4);
#pragma unroll
                for (int e = 0; e < 4; ++e)
                    VT[(seg + q * 4 + e) * VTSF + kk] = (_Float16)f[e];
            }
        }
        __syncthreads();

        v8f s[2];
#pragma unroll
        for (int nt = 0; nt < 2; ++nt)
#pragma unroll
            for (int r = 0; r < 8; ++r) s[nt][r] = 0.0f;
#pragma unroll
        for (int nt = 0; nt < 2; ++nt) {
            const int key = j * BNF + nt * 16 + l16;
            const int kb  = half ? 8 : 0;
#pragma unroll
            for (int dc = 0; dc < 2; ++dc) {
                const float* krow = K + bbase + (size_t)key * DHEAD + dc * 32 + kb;
                v16h bk;
#pragma unroll
                for (int q = 0; q < 4; ++q) {
                    v4f f = *(const v4f*)(krow + (q >> 1) * 16 + (q & 1) * 4);
#pragma unroll
                    for (int e = 0; e < 4; ++e) bk[q * 4 + e] = (_Float16)f[e];
                }
                s[nt] = __builtin_amdgcn_wmma_f32_16x16x32_f16(
                    false, aq[dc], false, bk, (short)0, s[nt], false, false);
            }
        }

        const bool needm = (j * BNF + BNF - 1) > q0;
        float sv[2][8];
#pragma unroll
        for (int nt = 0; nt < 2; ++nt)
#pragma unroll
            for (int r = 0; r < 8; ++r) {
                float x = s[nt][r] * 0.125f;
                if (needm) {
                    const int key = j * BNF + nt * 16 + l16;
                    const int row = q0 + r + half * 8;
                    if (key > row) x = NEG_BIG;
                }
                sv[nt][r] = x;
            }

        float mnew[8], alpha[8];
#pragma unroll
        for (int r = 0; r < 8; ++r) {
            float rm = fmaxf(sv[0][r], sv[1][r]);
#pragma unroll
            for (int off = 1; off < 16; off <<= 1)
                rm = fmaxf(rm, __shfl_xor(rm, off, 32));
            mnew[r]  = fmaxf(mr[r], rm);
            alpha[r] = __expf(mr[r] - mnew[r]);
            mr[r]    = mnew[r];
        }
#pragma unroll
        for (int r = 0; r < 8; ++r) {
            float p0 = __expf(sv[0][r] - mnew[r]);
            float p1 = __expf(sv[1][r] - mnew[r]);
            PT[(r + half * 8) * PSF + l16]      = (_Float16)p0;
            PT[(r + half * 8) * PSF + 16 + l16] = (_Float16)p1;
            lr[r] = alpha[r] * lr[r] + p0 + p1;   // per-lane partial
        }
#pragma unroll
        for (int t = 0; t < 4; ++t)
#pragma unroll
            for (int r = 0; r < 8; ++r) o[t][r] *= alpha[r];

        asm volatile("s_wait_dscnt 0" ::: "memory");

        v16h ap;
        {
            const int kb = half ? 8 : 0;
#pragma unroll
            for (int g = 0; g < 2; ++g) {
                v4h h0 = *(const v4h*)(PT + l16 * PSF + kb + g * 16);
                v4h h1 = *(const v4h*)(PT + l16 * PSF + kb + g * 16 + 4);
#pragma unroll
                for (int e = 0; e < 4; ++e) {
                    ap[g * 8 + e]     = h0[e];
                    ap[g * 8 + 4 + e] = h1[e];
                }
            }
        }
#pragma unroll
        for (int t = 0; t < 4; ++t) {
            const int d  = t * 16 + l16;
            const int kb = half ? 8 : 0;
            v16h bv;
#pragma unroll
            for (int q = 0; q < 4; ++q) {
                v4h h = *(const v4h*)(VT + d * VTSF + kb + (q >> 1) * 16 + (q & 1) * 4);
#pragma unroll
                for (int e = 0; e < 4; ++e) bv[q * 4 + e] = h[e];
            }
            o[t] = __builtin_amdgcn_wmma_f32_16x16x32_f16(
                false, ap, false, bv, (short)0, o[t], false, false);
        }
    }

#pragma unroll
    for (int r = 0; r < 8; ++r) {
        float ls = lr[r];
#pragma unroll
        for (int off = 1; off < 16; off <<= 1)
            ls += __shfl_xor(ls, off, 32);
        const float inv = 1.0f / ls;
        float* orow = Out + bbase + (size_t)(q0 + r + half * 8) * DHEAD + l16;
#pragma unroll
        for (int t = 0; t < 4; ++t)
            orow[t * 16] = o[t][r] * inv;
    }
}

// ============================ launcher ============================

extern "C" void kernel_launch(void* const* d_in, const int* in_sizes, int n_in,
                              void* d_out, int out_size, void* d_ws, size_t ws_size,
                              hipStream_t stream) {
    const float* Q = (const float*)d_in[0];
    const float* K = (const float*)d_in[1];
    const float* V = (const float*)d_in[2];
    float* Out = (float*)d_out;

    const int B = in_sizes[0] / (S_LEN * DHEAD);
    const size_t N = (size_t)B * S_LEN * DHEAD;
    const size_t need = N * 2 * sizeof(_Float16);   // Kh + Vt

    if (ws_size >= need) {
        _Float16* Kh = (_Float16*)d_ws;
        _Float16* Vt = Kh + N;
        hipLaunchKernelGGL(prep_k_kernel, dim3((unsigned)(N / 4 / 256)), dim3(256),
                           0, stream, K, Kh);
        hipLaunchKernelGGL(prep_vt_kernel, dim3(S_LEN / 64, B), dim3(256),
                           0, stream, V, Vt);
        hipLaunchKernelGGL(fa_fast_kernel, dim3(S_LEN / 128, B), dim3(256),
                           0, stream, Q, Kh, Vt, Out);
    } else {
        hipLaunchKernelGGL(fa_causal_fallback, dim3(S_LEN / BMF, B), dim3(128),
                           0, stream, Q, K, V, Out);
    }
}
